// PatchInpainting_74526272520668
// MI455X (gfx1250) — matrix operand order repaired and checked by reading, not compiled, not run on hardware
//
#include <hip/hip_runtime.h>
#include <hip/hip_bf16.h>
#include <float.h>

// ---------------------------------------------------------------------------
// Types for CDNA5 WMMA (wave32): v_wmma_f32_16x16x32_bf16
// ---------------------------------------------------------------------------
typedef __attribute__((ext_vector_type(16))) __bf16 v16bf;
typedef __attribute__((ext_vector_type(8)))  float  v8f;

#define NEGF (-3.402823466e+38f)

// Problem constants
#define B_    4
#define LQ    1024
#define LK    1024
#define EMB   2048
#define NH    8
#define DK    128
#define DV    768
#define HDK   (NH * DK)   // 1024
#define HDV   (NH * DV)   // 6144

// ---------------------------------------------------------------------------
// Weight transpose + fp32 -> bf16 convert:  W[K][N] -> Wt[N][K]
// 32x32 LDS tile, block = 256 threads
// ---------------------------------------------------------------------------
__global__ __launch_bounds__(256)
void transpose_cvt_kernel(const float* __restrict__ W, __bf16* __restrict__ Wt,
                          int K, int N) {
  __shared__ __bf16 tile[32][33];
  int n0 = blockIdx.x * 32;
  int k0 = blockIdx.y * 32;
  int tx = threadIdx.x & 31;
  int ty = threadIdx.x >> 5;          // 0..7
  #pragma unroll
  for (int r = ty; r < 32; r += 8)
    tile[r][tx] = (__bf16)W[(size_t)(k0 + r) * N + (n0 + tx)];
  __syncthreads();
  #pragma unroll
  for (int r = ty; r < 32; r += 8)
    Wt[(size_t)(n0 + r) * K + (k0 + tx)] = tile[tx][r];
}

// ---------------------------------------------------------------------------
// Generic WMMA GEMM:  C[M,N] (+batch) = A[M,K] * Bt[N,K]^T
//   A: fp32 or bf16 row-major; the 16x32 A tile is staged per-block in LDS via
//      CDNA5 async global->LDS (GLOBAL_LOAD_ASYNC_TO_LDS_B128 + s_wait_asynccnt),
//      then converted to bf16 while building the WMMA A fragment.
//   Bt: bf16, N-major (K contiguous) -> matches WMMA B-fragment layout;
//       streamed per-wave with global_prefetch one k-tile ahead.
//   Each wave computes TWO 16x16 tiles along N sharing one A fragment
//   (block = 4 waves = 128 output columns, 16 output rows).
//   EPI 0: fp32 out, scaled      (C[m*ldc+n] = acc*scale)
//   EPI 1: bf16 out, row-major   (C[m*ldc+n])
//   EPI 2: bf16 out, transposed  (C[n*ldc+m])
// Batch (z) decomposed as ib = z/Hn, ih = z%Hn.
// ---------------------------------------------------------------------------
template <typename AT, int EPI>
__global__ __launch_bounds__(128)
void gemm_wmma_kernel(const AT* __restrict__ A, size_t aB, size_t aH, int lda,
                      const __bf16* __restrict__ Bt, size_t bB, size_t bH, int ldb,
                      void* __restrict__ Cv, size_t cB, size_t cH, int ldc,
                      int K, float scale, int Hn) {
  __shared__ AT tileA[16 * 32];                 // 16 rows x 32 K (2KB f32 / 1KB bf16)

  const int z  = blockIdx.z;
  const int ib = z / Hn;
  const int ih = z - ib * Hn;
  const AT*     Ab = A  + aB * (size_t)ib + aH * (size_t)ih;
  const __bf16* Bb = Bt + bB * (size_t)ib + bH * (size_t)ih;

  const int tid    = threadIdx.x;
  const int lane   = tid & 31;
  const int wave   = tid >> 5;
  const int m_base = blockIdx.y * 16;
  const int n_base = blockIdx.x * 128 + wave * 32;

  // B fragments: lane L holds column N = L&15; K = {0..7,16..23} (L<16) / {8..15,24..31} (L>=16)
  const int bn   = n_base + (lane & 15);
  const int boff = (lane & 16) ? 8 : 0;
  const __bf16* brow0 = Bb + (size_t)bn * ldb + boff;
  const __bf16* brow1 = brow0 + (size_t)16 * ldb;

  // A fragment source in LDS: lane L reads row M = L&15; K-chunks
  // {0..7,16..23} (L<16) or {8..15,24..31} (L>=16), per ISA 16-bit A layout.
  const AT* arow_l = tileA + (lane & 15) * 32 + ((lane & 16) ? 8 : 0);

  // Async-copy geometry: one b128 (16B) per participating thread.
  constexpr int CPR = (32 * (int)sizeof(AT)) / 16;   // 16B chunks per A row
  constexpr int NTH = 16 * CPR;                      // threads needed (64 or 128)
  const unsigned lds_base = (unsigned)(unsigned long long)(uintptr_t)(void*)tileA;

  v8f acc0 = {}, acc1 = {};
  for (int k0 = 0; k0 < K; k0 += 32) {
    // ---- stage A tile [16][32] into LDS with CDNA5 async loads ----
    if (tid < NTH) {
      const int row = tid / CPR;
      const int ch  = tid - row * CPR;
      unsigned long long ga = (unsigned long long)(uintptr_t)
          ((const char*)(Ab + (size_t)(m_base + row) * lda + k0) + ch * 16);
      unsigned lo = lds_base + (unsigned)(tid * 16);
      asm volatile("global_load_async_to_lds_b128 %0, %1, off"
                   :: "v"(lo), "v"(ga) : "memory");
    }
    asm volatile("s_wait_asynccnt 0x0" ::: "memory");
    __syncthreads();

    // ---- build fragments ----
    v16bf a, b0, b1;
    #pragma unroll
    for (int i = 0; i < 8; ++i)  a[i] = (__bf16)arow_l[i];
    #pragma unroll
    for (int i = 8; i < 16; ++i) a[i] = (__bf16)arow_l[i + 8];
    #pragma unroll
    for (int i = 0; i < 16; ++i) { b0[i] = brow0[k0 + (i < 8 ? i : i + 8)]; b1[i] = brow1[k0 + (i < 8 ? i : i + 8)]; }
    if (k0 + 32 < K) {                       // prefetch next B k-tile
      __builtin_prefetch(brow0 + k0 + 32, 0, 1);
      __builtin_prefetch(brow1 + k0 + 32, 0, 1);
    }

    acc0 = __builtin_amdgcn_wmma_f32_16x16x32_bf16(
        false, a, false, b0, (short)0, acc0, false, false);
    acc1 = __builtin_amdgcn_wmma_f32_16x16x32_bf16(
        false, a, false, b1, (short)0, acc1, false, false);

    __syncthreads();                         // protect LDS tile before next store
  }

  // C fragment: acc[j] = (M = m_base + j + 8*(lane>=16), N = tile_n + lane&15)
  const int cn0 = n_base + (lane & 15);
  const int cn1 = cn0 + 16;
  const int cm  = m_base + ((lane & 16) ? 8 : 0);
  if (EPI == 0) {
    float* C = (float*)Cv + cB * (size_t)ib + cH * (size_t)ih;
    #pragma unroll
    for (int j = 0; j < 8; ++j) {
      C[(size_t)(cm + j) * ldc + cn0] = acc0[j] * scale;
      C[(size_t)(cm + j) * ldc + cn1] = acc1[j] * scale;
    }
  } else if (EPI == 1) {
    __bf16* C = (__bf16*)Cv + cB * (size_t)ib + cH * (size_t)ih;
    #pragma unroll
    for (int j = 0; j < 8; ++j) {
      C[(size_t)(cm + j) * ldc + cn0] = (__bf16)acc0[j];
      C[(size_t)(cm + j) * ldc + cn1] = (__bf16)acc1[j];
    }
  } else {
    __bf16* C = (__bf16*)Cv + cB * (size_t)ib + cH * (size_t)ih;
    #pragma unroll
    for (int j = 0; j < 8; ++j) {
      C[(size_t)cn0 * ldc + (cm + j)] = (__bf16)acc0[j];
      C[(size_t)cn1 * ldc + (cm + j)] = (__bf16)acc1[j];
    }
  }
}

// ---------------------------------------------------------------------------
// Top-k(32) restriction + softmax, in place on the attn slice of d_out.
// One wave32 per row of 1024 logits (block = 256 = 8 rows).
// Rows indexed as ((b*8+h)*1024+q). Applies post_softmax_mask (==0 -> NEG),
// then for rows with query_mask>0.5 keeps only the top-32 logits.
// ---------------------------------------------------------------------------
__global__ __launch_bounds__(256)
void topk_softmax_kernel(float* __restrict__ attn,
                         const float* __restrict__ mask,   // [B,1,Lq,Lk]
                         const float* __restrict__ qmask)  // [B,Lq]
{
  const int lane = threadIdx.x & 31;
  const int row  = blockIdx.x * 8 + (threadIdx.x >> 5);
  const int b = row >> 13;
  const int q = row & 1023;

  float* rowp = attn + (size_t)row * LK;
  const float* mrow = mask + ((size_t)b * LQ + q) * LK;

  float vals[32];
  #pragma unroll
  for (int j = 0; j < 32; ++j) {
    int c = lane + (j << 5);
    float v = rowp[c];
    if (mrow[c] == 0.0f) v = NEGF;
    vals[j] = v;
  }

  const bool restrictq = qmask[b * LQ + q] > 0.5f;  // uniform per wave
  if (restrictq) {
    float work[32];
    #pragma unroll
    for (int j = 0; j < 32; ++j) work[j] = vals[j];
    float thresh = NEGF;
    for (int t = 0; t < 32; ++t) {
      float lm = NEGF;
      #pragma unroll
      for (int j = 0; j < 32; ++j) lm = fmaxf(lm, work[j]);
      float gm = lm;
      #pragma unroll
      for (int o = 16; o > 0; o >>= 1) gm = fmaxf(gm, __shfl_xor(gm, o, 32));
      unsigned long long bal = __ballot(lm == gm);
      int src = (int)__ffsll(bal) - 1;
      if (lane == src) {           // remove exactly one instance of the max
        bool done = false;
        #pragma unroll
        for (int j = 0; j < 32; ++j) {
          if (!done && work[j] == gm) { work[j] = NEGF; done = true; }
        }
      }
      thresh = gm;                 // after 32 extractions: 32nd-largest
    }
    #pragma unroll
    for (int j = 0; j < 32; ++j)
      if (vals[j] < thresh) vals[j] = NEGF;
  }

  // numerically stable softmax
  float rmax = NEGF;
  #pragma unroll
  for (int j = 0; j < 32; ++j) rmax = fmaxf(rmax, vals[j]);
  #pragma unroll
  for (int o = 16; o > 0; o >>= 1) rmax = fmaxf(rmax, __shfl_xor(rmax, o, 32));
  float s = 0.0f;
  #pragma unroll
  for (int j = 0; j < 32; ++j) { vals[j] = __expf(vals[j] - rmax); s += vals[j]; }
  #pragma unroll
  for (int o = 16; o > 0; o >>= 1) s += __shfl_xor(s, o, 32);
  const float inv = 1.0f / s;
  #pragma unroll
  for (int j = 0; j < 32; ++j) rowp[lane + (j << 5)] = vals[j] * inv;
}

// ---------------------------------------------------------------------------
// Host launcher
// ---------------------------------------------------------------------------
extern "C" void kernel_launch(void* const* d_in, const int* in_sizes, int n_in,
                              void* d_out, int out_size, void* d_ws, size_t ws_size,
                              hipStream_t stream) {
  const float* q     = (const float*)d_in[0];
  const float* k     = (const float*)d_in[1];
  const float* v     = (const float*)d_in[2];
  const float* mask  = (const float*)d_in[3];
  const float* qmask = (const float*)d_in[4];
  const float* w_qs  = (const float*)d_in[5];
  const float* w_ks  = (const float*)d_in[6];
  const float* w_vs  = (const float*)d_in[7];
  const float* fc    = (const float*)d_in[8];

  float* out  = (float*)d_out;                         // [4,1024,768]
  float* attn = out + (size_t)B_ * LQ * DV;            // [4,8,1024,1024]

  // Workspace layout (bf16 regions)
  char* ws = (char*)d_ws;
  __bf16* w_qs_t = (__bf16*)ws;                 ws += (size_t)HDK * EMB * 2;   // [1024][2048]
  __bf16* w_ks_t = (__bf16*)ws;                 ws += (size_t)HDK * EMB * 2;   // [1024][2048]
  __bf16* w_vs_t = (__bf16*)ws;                 ws += (size_t)HDV * DV  * 2;   // [6144][768]
  __bf16* fc_t   = (__bf16*)ws;                 ws += (size_t)DV  * HDV * 2;   // [768][6144]
  __bf16* qp     = (__bf16*)ws;                 ws += (size_t)B_ * LQ * HDK * 2; // [4096][1024]
  __bf16* kp     = (__bf16*)ws;                 ws += (size_t)B_ * LK * HDK * 2; // [4096][1024]
  __bf16* vp_t   = (__bf16*)ws;                 ws += (size_t)B_ * NH * DV * LK * 2; // [b][h][dv][key]
  __bf16* mixed  = (__bf16*)ws;                 /* [4096][6144] */

  // 1) weight transpose + convert to bf16 [N][K]
  transpose_cvt_kernel<<<dim3(HDK / 32, EMB / 32), 256, 0, stream>>>(w_qs, w_qs_t, EMB, HDK);
  transpose_cvt_kernel<<<dim3(HDK / 32, EMB / 32), 256, 0, stream>>>(w_ks, w_ks_t, EMB, HDK);
  transpose_cvt_kernel<<<dim3(HDV / 32, DV  / 32), 256, 0, stream>>>(w_vs, w_vs_t, DV,  HDV);
  transpose_cvt_kernel<<<dim3(DV  / 32, HDV / 32), 256, 0, stream>>>(fc,   fc_t,   HDV, DV);

  // 2) qp = q @ w_qs  -> bf16 [4096][1024]   (1/sqrt(dk) folded into logits)
  gemm_wmma_kernel<float, 1><<<dim3(HDK / 128, (B_ * LQ) / 16, 1), 128, 0, stream>>>(
      q, 0, 0, EMB, w_qs_t, 0, 0, EMB, qp, 0, 0, HDK, EMB, 1.0f, 1);
  // 3) kp = k @ w_ks  -> bf16 [4096][1024]
  gemm_wmma_kernel<float, 1><<<dim3(HDK / 128, (B_ * LK) / 16, 1), 128, 0, stream>>>(
      k, 0, 0, EMB, w_ks_t, 0, 0, EMB, kp, 0, 0, HDK, EMB, 1.0f, 1);
  // 4) vp = v @ w_vs  -> bf16 transposed  vp_t[b][h][dv][key]
  gemm_wmma_kernel<float, 2><<<dim3(HDV / 128, LK / 16, B_), 128, 0, stream>>>(
      v, (size_t)LK * DV, 0, DV, w_vs_t, 0, 0, DV,
      vp_t, (size_t)HDV * LK, 0, /*ldc = M len*/ LK, DV, 1.0f, 1);

  // 5) logits[b,h] = (qp/sqrt(dk)) @ kp^T -> fp32 into attn slice of d_out
  const float inv_sqrt_dk = 0.08838834764831845f;  // 1/sqrt(128), TEMP=1
  gemm_wmma_kernel<__bf16, 0><<<dim3(LK / 128, LQ / 16, B_ * NH), 128, 0, stream>>>(
      qp, (size_t)LQ * HDK, DK, HDK,
      kp, (size_t)LK * HDK, DK, HDK,
      attn, (size_t)NH * LQ * LK, (size_t)LQ * LK, LK,
      DK, inv_sqrt_dk, NH);

  // 6) mask + top-k(32) restriction + softmax, in place
  topk_softmax_kernel<<<dim3((B_ * NH * LQ) / 8), 256, 0, stream>>>(attn, mask, qmask);

  // 7) mixed[b][q][h*768+dv] = attn @ vp  -> bf16 [4096][6144]
  gemm_wmma_kernel<float, 1><<<dim3(DV / 128, LQ / 16, B_ * NH), 128, 0, stream>>>(
      attn, (size_t)NH * LQ * LK, (size_t)LQ * LK, LK,
      vp_t, (size_t)NH * DV * LK, (size_t)DV * LK, LK,
      mixed, (size_t)LQ * HDV, DV, HDV,
      LK, 1.0f, NH);

  // 8) out = mixed @ fc  -> fp32 [4096][768]
  gemm_wmma_kernel<__bf16, 0><<<dim3(DV / 128, (B_ * LQ) / 16, 1), 128, 0, stream>>>(
      mixed, 0, 0, HDV, fc_t, 0, 0, HDV, out, 0, 0, DV, HDV, 1.0f, 1);

  (void)in_sizes; (void)n_in; (void)out_size; (void)ws_size;
}
